// ShortLongMixLayer_18081812316204
// MI455X (gfx1250) — matrix operand order repaired and checked by reading, not compiled, not run on hardware
//
#include <hip/hip_runtime.h>
#include <hip/hip_bf16.h>

// ---------------------------------------------------------------------------
// ShortLongMixLayer for MI455X (gfx1250, wave32, WMMA).
// - All GEMM-shaped math (node projections, RBF edge filters, 3x3x3 conv) via
//   v_wmma_f32_16x16x32_bf16.
// - Edge-attr tiles staged with global_load_async_to_lds_b128 (ASYNCcnt path).
// - Scatter via no-return global_atomic_add_f32 into L2-resident accumulators.
// ---------------------------------------------------------------------------

typedef __attribute__((ext_vector_type(16))) __bf16 v16bf;
typedef __attribute__((ext_vector_type(8)))  float  v8f;

constexpr int kNA   = 50000;
constexpr int kNM   = 16384;
constexpr int kH    = 128;
constexpr int kRBF  = 64;
constexpr int kEAA  = 800000;
constexpr int kEAM  = 400000;
constexpr int kEMA  = 400000;
constexpr float kEPS = 1e-5f;

// ---------------- wave helpers ----------------
__device__ __forceinline__ float wave_sum32(float v) {
#pragma unroll
  for (int m = 16; m >= 1; m >>= 1) v += __shfl_xor(v, m, 32);
  return v;
}

__device__ __forceinline__ v8f wmma_bf16(v16bf a, v16bf b, v8f c) {
  return __builtin_amdgcn_wmma_f32_16x16x32_bf16(false, a, false, b,
                                                 (short)0, c, false, false);
}

__device__ __forceinline__ unsigned pack_bf16(float x, float y) {
  __bf16 lo = (__bf16)x, hi = (__bf16)y;
  unsigned short ul = __builtin_bit_cast(unsigned short, lo);
  unsigned short uh = __builtin_bit_cast(unsigned short, hi);
  return ((unsigned)uh << 16) | (unsigned)ul;
}

// A-fragment (16x32 bf16) from an LDS row-major bf16 tile, per ISA 7.12.2:
//   lane l holds row M = l&15; element j maps to K = (j>>3)*16 + (j&7) + 8*(l>>4).
__device__ __forceinline__ v16bf load_a_frag(const __bf16* tile, int stride,
                                             int k0, int lane) {
  v16bf a;
  const __bf16* p = tile + (lane & 15) * stride + k0 + ((lane >> 4) << 3);
#pragma unroll
  for (int j = 0; j < 16; ++j) {
    a[j] = p[((j >> 3) << 4) + (j & 7)];
  }
  return a;
}

// Same mapping but source tile is f32 in LDS (async-copied raw edge attrs).
__device__ __forceinline__ v16bf load_a_frag_f32(const float* tile, int stride,
                                                 int k0, int lane) {
  v16bf a;
  const float* p = tile + (lane & 15) * stride + k0 + ((lane >> 4) << 3);
#pragma unroll
  for (int j = 0; j < 16; ++j) {
    a[j] = (__bf16)p[((j >> 3) << 4) + (j & 7)];
  }
  return a;
}

// ---------------- weight fragment packing ----------------
// B (32x16 slice, bf16): VGPR-half j = K (within 32-slice) = (j>>3)*16 + (j&7) + 8 for lanes 16-31,
// lane&15 = N. Packed so each lane reads 16 contiguous halfs (32 bytes).
// dst index = ((t*(K/32) + s)*32 + lane)*16 + j
__global__ void pack_weight_frag(const float* __restrict__ S, int K,
                                 __bf16* __restrict__ dst) {
  int idx = blockIdx.x * 256 + threadIdx.x;
  int ks = K >> 5;
  int total = 8 * ks * 512;
  if (idx >= total) return;
  int j    = idx & 15;
  int lane = (idx >> 4) & 31;
  int s    = (idx >> 9) % ks;
  int t    = (idx >> 9) / ks;
  int k = 32 * s + ((j >> 3) << 4) + (j & 7) + ((lane >> 4) << 3);
  int n = 16 * t + (lane & 15);
  dst[idx] = (__bf16)S[k * kH + n];
}

// conv_k is (O=128, I=128, 3,3,3) row-major; per tap build B[k=i][n=o].
__global__ void pack_conv_frag(const float* __restrict__ ck,
                               __bf16* __restrict__ dst) {
  int idx = blockIdx.x * 256 + threadIdx.x;
  if (idx >= 27 * 16384) return;
  int tap = idx / 16384;
  int r   = idx & 16383;
  int j    = r & 15;
  int lane = (r >> 4) & 31;
  int s    = (r >> 9) & 3;
  int t    = r >> 11;
  int i = 32 * s + ((j >> 3) << 4) + (j & 7) + ((lane >> 4) << 3);
  int o = 16 * t + (lane & 15);
  dst[idx] = (__bf16)ck[(o * kH + i) * 27 + tap];
}

__global__ void zero_f32(float* __restrict__ p, int n) {
  int i = blockIdx.x * 256 + threadIdx.x;
  if (i < n) p[i] = 0.f;
}

// ---------------- LayerNorm -> bf16 rows (wave per row) ----------------
__global__ void ln_rows_bf16(const float* __restrict__ x,
                             const float* __restrict__ g,
                             const float* __restrict__ b,
                             __bf16* __restrict__ out, int nrows) {
  int row  = blockIdx.x * 8 + (threadIdx.x >> 5);
  int lane = threadIdx.x & 31;
  if (row >= nrows) return;
  const float* xr = x + (size_t)row * kH;
  float v[4];
  float sm = 0.f;
#pragma unroll
  for (int c = 0; c < 4; ++c) { v[c] = xr[lane + 32 * c]; sm += v[c]; }
  float mu = wave_sum32(sm) * (1.f / kH);
  float q = 0.f;
#pragma unroll
  for (int c = 0; c < 4; ++c) { float d = v[c] - mu; q += d * d; }
  float rs = rsqrtf(wave_sum32(q) * (1.f / kH) + kEPS);
#pragma unroll
  for (int c = 0; c < 4; ++c) {
    int n = lane + 32 * c;
    out[(size_t)row * kH + n] = (__bf16)((v[c] - mu) * rs * g[n] + b[n]);
  }
}

// Stage one row (optionally LayerNormed) into an LDS bf16 row, packed 32-bit
// writes. `valid` is wave-uniform.
__device__ __forceinline__ void stage_row_ln(const float* __restrict__ xr,
                                             bool valid,
                                             const float* __restrict__ g,
                                             const float* __restrict__ bta,
                                             int do_ln, __bf16* dstrow,
                                             int lane) {
  float v0 = 0.f, v1 = 0.f, v2 = 0.f, v3 = 0.f;
  if (valid) {
    float2 p0 = ((const float2*)xr)[lane];        // cols 2l, 2l+1
    float2 p1 = ((const float2*)xr)[32 + lane];   // cols 64+2l, 65+2l
    v0 = p0.x; v1 = p0.y; v2 = p1.x; v3 = p1.y;
    if (do_ln) {
      float mu = wave_sum32(v0 + v1 + v2 + v3) * (1.f / kH);
      float d0 = v0 - mu, d1 = v1 - mu, d2 = v2 - mu, d3 = v3 - mu;
      float rs = rsqrtf(wave_sum32(d0 * d0 + d1 * d1 + d2 * d2 + d3 * d3) *
                            (1.f / kH) + kEPS);
      int n0 = 2 * lane, n2 = 64 + 2 * lane;
      v0 = d0 * rs * g[n0]     + bta[n0];
      v1 = d1 * rs * g[n0 + 1] + bta[n0 + 1];
      v2 = d2 * rs * g[n2]     + bta[n2];
      v3 = d3 * rs * g[n2 + 1] + bta[n2 + 1];
    }
  }
  ((unsigned*)dstrow)[lane]      = pack_bf16(v0, v1);
  ((unsigned*)dstrow)[32 + lane] = pack_bf16(v2, v3);
}

// ---------------- fused (optional LN) -> x @ W  (K=128, N=128) ----------------
// Block = 256 threads = 8 waves; each wave produces 16 rows of z (bf16).
__global__ void node_gemm(const float* __restrict__ x, int nrows,
                          const float* __restrict__ g,
                          const float* __restrict__ bta, int do_ln,
                          const __bf16* __restrict__ wfrag,
                          __bf16* __restrict__ z) {
  __shared__ __bf16 tile[8][16][kH];
  int w = threadIdx.x >> 5, lane = threadIdx.x & 31;
  int row0 = blockIdx.x * 128 + w * 16;
  bool full = (row0 + 16 <= nrows);

  if (full) {
#pragma unroll 4
    for (int rr = 0; rr < 16; ++rr) {
      stage_row_ln(x + (size_t)(row0 + rr) * kH, true, g, bta, do_ln,
                   &tile[w][rr][0], lane);
    }
  } else {
    for (int rr = 0; rr < 16; ++rr) {
      stage_row_ln(x + (size_t)(row0 + rr) * kH, (row0 + rr) < nrows, g, bta,
                   do_ln, &tile[w][rr][0], lane);
    }
  }

  v16bf a[4];
#pragma unroll
  for (int s = 0; s < 4; ++s) a[s] = load_a_frag(&tile[w][0][0], kH, 32 * s, lane);

  v8f acc[8];
#pragma unroll
  for (int t = 0; t < 8; ++t) {
    v8f c = {};
#pragma unroll
    for (int s = 0; s < 4; ++s) {
      v16bf bf = *(const v16bf*)(wfrag + (((t * 4 + s) * 32 + lane) << 4));
      c = wmma_bf16(a[s], bf, c);
    }
    acc[t] = c;
  }

  int mb = (lane >> 4) << 3;
  int nc = lane & 15;
  if (full) {
#pragma unroll
    for (int t = 0; t < 8; ++t) {
      int n = 16 * t + nc;
#pragma unroll
      for (int r = 0; r < 8; ++r) {
        z[(size_t)(row0 + mb + r) * kH + n] = (__bf16)acc[t][r];
      }
    }
  } else {
#pragma unroll
    for (int t = 0; t < 8; ++t) {
      int n = 16 * t + nc;
      for (int r = 0; r < 8; ++r) {
        int row = row0 + mb + r;
        if (row < nrows) z[(size_t)row * kH + n] = (__bf16)acc[t][r];
      }
    }
  }
}

// ---------------- 3D conv over mesh grid as 27 accumulated WMMA GEMMs --------
__global__ void conv3d_mesh(const __bf16* __restrict__ m1,
                            const __bf16* __restrict__ cfrag,
                            const float* __restrict__ bias,
                            float* __restrict__ m2) {
  __shared__ __bf16 tile[8][16][kH];
  int w = threadIdx.x >> 5, lane = threadIdx.x & 31;
  int v0 = blockIdx.x * 128 + w * 16;          // 16 consecutive voxels per wave
  int bb  = v0 >> 9;
  int gx  = (v0 >> 6) & 7;
  int gy0 = (v0 >> 3) & 7;                     // even; rows 0-7 -> gy0, 8-15 -> gy0+1

  v8f acc[8];
#pragma unroll
  for (int t = 0; t < 8; ++t) { v8f zz = {}; acc[t] = zz; }

  for (int tap = 0; tap < 27; ++tap) {
    int kd = tap / 9 - 1;
    int kh = (tap / 3) % 3 - 1;
    int kw = tap % 3 - 1;
    int sx = gx + kd;
    bool xok = (unsigned)sx < 8u;
    for (int rr = 0; rr < 16; ++rr) {
      int sy = gy0 + (rr >> 3) + kh;
      int sz = (rr & 7) + kw;
      uint2 val = make_uint2(0u, 0u);
      if (xok && (unsigned)sy < 8u && (unsigned)sz < 8u) {
        int sv = ((bb * 8 + sx) * 8 + sy) * 8 + sz;
        val = ((const uint2*)(m1 + (size_t)sv * kH))[lane];
      }
      ((uint2*)&tile[w][rr][0])[lane] = val;   // per-wave LDS region, DS in-order
    }
    v16bf a[4];
#pragma unroll
    for (int s = 0; s < 4; ++s) a[s] = load_a_frag(&tile[w][0][0], kH, 32 * s, lane);
    const __bf16* fb = cfrag + tap * 16384;
#pragma unroll
    for (int t = 0; t < 8; ++t) {
#pragma unroll
      for (int s = 0; s < 4; ++s) {
        v16bf bf = *(const v16bf*)(fb + (((t * 4 + s) * 32 + lane) << 4));
        acc[t] = wmma_bf16(a[s], bf, acc[t]);
      }
    }
  }

  int mb = (lane >> 4) << 3;
#pragma unroll
  for (int t = 0; t < 8; ++t) {
    int n = 16 * t + (lane & 15);
    float bv = bias[n];
#pragma unroll
    for (int r = 0; r < 8; ++r) {
      m2[(size_t)(v0 + mb + r) * kH + n] = acc[t][r] + bv;
    }
  }
}

// ---------------- fused edge message pass -----------------------------------
// Per wave: 16 edges. Edge-attr tile (16x64 f32 = 4KB, contiguous in memory)
// is DMA'd into LDS with global_load_async_to_lds_b128 (ASYNCcnt), then
// filt = ea @ We via WMMA (K=64), msg = filt * z[src] * w, scatter with
// global_atomic_add_f32 (no return -> STOREcnt, resolves in L2).
__global__ void edge_mp(const int* __restrict__ src, const int* __restrict__ dst,
                        const float* __restrict__ ew, const float* __restrict__ ea,
                        int E, const __bf16* __restrict__ wefrag,
                        const __bf16* __restrict__ z, float* __restrict__ out) {
  __shared__ float tilef[8][16][kRBF];
  int w = threadIdx.x >> 5, lane = threadIdx.x & 31;
  int e0 = blockIdx.x * 128 + w * 16;
  if (e0 >= E) return;

  // stream-prefetch next tile of edge attrs (global_prefetch_b8)
  if (e0 + 128 < E) __builtin_prefetch(ea + (size_t)(e0 + 128) * kRBF, 0, 1);

  // Async copy 4KB tile: 8 x B128, each lane moves 16B per op.
  {
    unsigned long long gbase =
        (unsigned long long)(ea + (size_t)e0 * kRBF) + (unsigned)(lane * 16);
    unsigned lbase = (unsigned)(size_t)(&tilef[w][0][0]) + (unsigned)(lane * 16);
#pragma unroll
    for (int it = 0; it < 8; ++it) {
      unsigned long long ga = gbase + (unsigned)(it * 512);
      unsigned la = lbase + (unsigned)(it * 512);
      asm volatile("global_load_async_to_lds_b128 %0, %1, off"
                   :: "v"(la), "v"(ga) : "memory");
    }
    asm volatile("s_wait_asynccnt 0x0" ::: "memory");
  }

  v16bf a0 = load_a_frag_f32(&tilef[w][0][0], kRBF, 0, lane);
  v16bf a1 = load_a_frag_f32(&tilef[w][0][0], kRBF, 32, lane);

  int mb = (lane >> 4) << 3;
  int srow[8], drow[8];
  float ew8[8];
#pragma unroll
  for (int r = 0; r < 8; ++r) {
    int e = e0 + mb + r;
    srow[r] = src[e] * kH;
    drow[r] = dst[e] * kH;
    ew8[r]  = ew[e];
  }

#pragma unroll
  for (int t = 0; t < 8; ++t) {
    v16bf b0 = *(const v16bf*)(wefrag + (((t * 2 + 0) * 32 + lane) << 4));
    v16bf b1 = *(const v16bf*)(wefrag + (((t * 2 + 1) * 32 + lane) << 4));
    v8f c = {};
    c = wmma_bf16(a0, b0, c);
    c = wmma_bf16(a1, b1, c);
    int n = 16 * t + (lane & 15);
#pragma unroll
    for (int r = 0; r < 8; ++r) {
      float zv  = (float)z[(size_t)(srow[r] + n)];
      float val = c[r] * zv * ew8[r];
      __hip_atomic_fetch_add(&out[(size_t)(drow[r] + n)], val,
                             __ATOMIC_RELAXED, __HIP_MEMORY_SCOPE_AGENT);
    }
  }
}

// ---------------- finalize: out = base + LN(msg)*g+b + delta ----------------
__global__ void finalize_ln_add(const float* __restrict__ base,
                                const float* __restrict__ msg,
                                const float* __restrict__ delta,
                                const float* __restrict__ g,
                                const float* __restrict__ bta,
                                float* __restrict__ out, int nrows) {
  int row  = blockIdx.x * 8 + (threadIdx.x >> 5);
  int lane = threadIdx.x & 31;
  if (row >= nrows) return;
  size_t off = (size_t)row * kH;
  float v[4];
  float sm = 0.f;
#pragma unroll
  for (int c = 0; c < 4; ++c) { v[c] = msg[off + lane + 32 * c]; sm += v[c]; }
  float mu = wave_sum32(sm) * (1.f / kH);
  float q = 0.f;
#pragma unroll
  for (int c = 0; c < 4; ++c) { float d = v[c] - mu; q += d * d; }
  float rs = rsqrtf(wave_sum32(q) * (1.f / kH) + kEPS);
#pragma unroll
  for (int c = 0; c < 4; ++c) {
    int n = lane + 32 * c;
    out[off + n] = base[off + n] + ((v[c] - mu) * rs * g[n] + bta[n]) + delta[off + n];
  }
}

// ---------------------------------------------------------------------------
extern "C" void kernel_launch(void* const* d_in, const int* in_sizes, int n_in,
                              void* d_out, int out_size, void* d_ws, size_t ws_size,
                              hipStream_t stream) {
  const float* a_x      = (const float*)d_in[0];
  const float* m_x      = (const float*)d_in[1];
  const int*   a2a_idx  = (const int*)d_in[2];
  const int*   a2a_src  = a2a_idx;
  const int*   a2a_dst  = a2a_idx + kEAA;
  const int*   a2m_src  = (const int*)d_in[3];
  const int*   a2m_dst  = (const int*)d_in[4];
  const int*   m2a_src  = (const int*)d_in[5];
  const int*   m2a_dst  = (const int*)d_in[6];
  const float* a2a_w    = (const float*)d_in[7];
  const float* a2m_w    = (const float*)d_in[8];
  const float* m2a_w    = (const float*)d_in[9];
  const float* a2a_attr = (const float*)d_in[10];
  const float* a2m_attr = (const float*)d_in[11];
  const float* m2a_attr = (const float*)d_in[12];
  const float* W_short  = (const float*)d_in[13];
  const float* We_short = (const float*)d_in[14];
  const float* W_a2m    = (const float*)d_in[15];
  const float* We_a2m   = (const float*)d_in[16];
  const float* W_m2a    = (const float*)d_in[17];
  const float* We_m2a   = (const float*)d_in[18];
  const float* conv_k   = (const float*)d_in[19];
  const float* conv_b   = (const float*)d_in[20];
  const float* g_short  = (const float*)d_in[21];
  const float* b_short  = (const float*)d_in[22];
  const float* g_long   = (const float*)d_in[23];
  const float* b_long   = (const float*)d_in[24];
  const float* g_a2m    = (const float*)d_in[25];
  const float* b_a2m    = (const float*)d_in[26];
  const float* g_m2a    = (const float*)d_in[27];
  const float* b_m2a    = (const float*)d_in[28];

  char* ws = (char*)d_ws;
  size_t off = 0;
  auto alloc = [&](size_t bytes) -> void* {
    void* p = ws + off;
    off = (off + bytes + 255) & ~(size_t)255;
    return p;
  };

  __bf16* fW_short  = (__bf16*)alloc((size_t)kH * kH * 2);
  __bf16* fW_a2m    = (__bf16*)alloc((size_t)kH * kH * 2);
  __bf16* fW_m2a    = (__bf16*)alloc((size_t)kH * kH * 2);
  __bf16* fWe_short = (__bf16*)alloc((size_t)kRBF * kH * 2);
  __bf16* fWe_a2m   = (__bf16*)alloc((size_t)kRBF * kH * 2);
  __bf16* fWe_m2a   = (__bf16*)alloc((size_t)kRBF * kH * 2);
  __bf16* fconv     = (__bf16*)alloc((size_t)27 * kH * kH * 2);
  __bf16* m1        = (__bf16*)alloc((size_t)kNM * kH * 2);
  __bf16* z_short   = (__bf16*)alloc((size_t)kNA * kH * 2);
  __bf16* z_a2m     = (__bf16*)alloc((size_t)kNA * kH * 2);
  __bf16* z_m2a     = (__bf16*)alloc((size_t)kNM * kH * 2);
  float*  msg_a     = (float*)alloc((size_t)kNA * kH * 4);   // a2 accumulator
  float*  msg_m     = (float*)alloc((size_t)kNM * kH * 4);   // a2m messages (mesh)
  float*  msg_a2    = (float*)alloc((size_t)kNA * kH * 4);   // m2a messages (atoms)
  float*  m2        = (float*)alloc((size_t)kNM * kH * 4);   // conv output

  dim3 blk(256);

  // --- pack weights into WMMA B-fragment order ---
  pack_weight_frag<<<64, blk, 0, stream>>>(W_short, kH, fW_short);
  pack_weight_frag<<<64, blk, 0, stream>>>(W_a2m,   kH, fW_a2m);
  pack_weight_frag<<<64, blk, 0, stream>>>(W_m2a,   kH, fW_m2a);
  pack_weight_frag<<<32, blk, 0, stream>>>(We_short, kRBF, fWe_short);
  pack_weight_frag<<<32, blk, 0, stream>>>(We_a2m,   kRBF, fWe_a2m);
  pack_weight_frag<<<32, blk, 0, stream>>>(We_m2a,   kRBF, fWe_m2a);
  pack_conv_frag<<<(27 * 16384 + 255) / 256, blk, 0, stream>>>(conv_k, fconv);

  // --- stage 1: LN branches ---
  ln_rows_bf16<<<(kNM + 7) / 8, blk, 0, stream>>>(m_x, g_long, b_long, m1, kNM);
  node_gemm<<<(kNA + 127) / 128, blk, 0, stream>>>(a_x, kNA, g_short, b_short, 1,
                                                   fW_short, z_short);

  // --- a2a message pass -> msg_a (== a2) ---
  zero_f32<<<(kNA * kH + 255) / 256, blk, 0, stream>>>(msg_a, kNA * kH);
  edge_mp<<<kEAA / 128, blk, 0, stream>>>(a2a_src, a2a_dst, a2a_w, a2a_attr,
                                          kEAA, fWe_short, z_short, msg_a);

  // --- long-range mesh conv -> m2 ---
  conv3d_mesh<<<kNM / 128, blk, 0, stream>>>(m1, fconv, conv_b, m2);

  // --- cross projections ---
  node_gemm<<<(kNA + 127) / 128, blk, 0, stream>>>(msg_a, kNA, nullptr, nullptr, 0,
                                                   fW_a2m, z_a2m);
  node_gemm<<<kNM / 128, blk, 0, stream>>>(m2, kNM, nullptr, nullptr, 0,
                                           fW_m2a, z_m2a);

  // --- cross message passes ---
  zero_f32<<<(kNM * kH + 255) / 256, blk, 0, stream>>>(msg_m, kNM * kH);
  zero_f32<<<(kNA * kH + 255) / 256, blk, 0, stream>>>(msg_a2, kNA * kH);
  edge_mp<<<kEAM / 128, blk, 0, stream>>>(a2m_src, a2m_dst, a2m_w, a2m_attr,
                                          kEAM, fWe_a2m, z_a2m, msg_m);
  edge_mp<<<kEMA / 128, blk, 0, stream>>>(m2a_src, m2a_dst, m2a_w, m2a_attr,
                                          kEMA, fWe_m2a, z_m2a, msg_a2);

  // --- finalize: out_a then out_m (concatenated in d_out) ---
  float* out_a = (float*)d_out;
  float* out_m = out_a + (size_t)kNA * kH;
  finalize_ln_add<<<(kNA + 7) / 8, blk, 0, stream>>>(msg_a, msg_a2, a_x,
                                                     g_m2a, b_m2a, out_a, kNA);
  finalize_ln_add<<<kNM / 8, blk, 0, stream>>>(m2, msg_m, m_x,
                                               g_a2m, b_a2m, out_m, kNM);
}
